// OIMLossSafe_51470888075899
// MI455X (gfx1250) — compile-verified, matrix operand order checked by reading, not run on hardware
//
#include <hip/hip_runtime.h>
#include <math.h>

#define NUM_FEATURES 256
#define NUM_PIDS     50000
#define CQ_SIZE      5000
#define TOTAL_COLS   (NUM_PIDS + CQ_SIZE)   // 55000
#define BATCH        2048
#define OIM_SCALAR   30.0f
#define EPSV         1e-12f
#define LOG2E        1.4426950408889634f

// GEMM blocking: 64x128 output per 256-thread block, K staged in 2 chunks of 128
#define TMB 64      // x rows per block
#define TNB 128     // W rows (logit columns) per block
#define KDIM 256
#define KC  128     // K chunk staged in LDS
#define LDSC 136    // padded LDS row stride (bf16): 272B = 68 DW -> 4-bank rotation/row

typedef __attribute__((ext_vector_type(8)))  float   v8f;
typedef __attribute__((ext_vector_type(8)))  __bf16  v8bf;
typedef __attribute__((ext_vector_type(16))) __bf16  v16bf;

// ---------------------------------------------------------------- utilities
__device__ __forceinline__ float wave_sum32(float s) {
    s += __shfl_xor(s, 1, 32);
    s += __shfl_xor(s, 2, 32);
    s += __shfl_xor(s, 4, 32);
    s += __shfl_xor(s, 8, 32);
    s += __shfl_xor(s, 16, 32);
    return s;
}

// 16-bit A/B 16x32 fragment: lanes0-15 -> K0-7|K16-23, lanes16-31 -> K8-15|K24-31
__device__ __forceinline__ v16bf load_frag(const __bf16* rowPtr, int cLo) {
    v8bf lo = *(const v8bf*)(rowPtr + cLo);
    v8bf hi = *(const v8bf*)(rowPtr + cLo + 16);
    return __builtin_shufflevector(lo, hi, 0,1,2,3,4,5,6,7,8,9,10,11,12,13,14,15);
}

// ------------------------------------------------- 1. L2-normalize x, zero S
__global__ __launch_bounds__(256) void normalize_kernel(
    const float* __restrict__ inp, float* __restrict__ xn, float* __restrict__ S) {
    int row  = blockIdx.x * 8 + (threadIdx.x >> 5);
    int lane = threadIdx.x & 31;
    const float4* src = (const float4*)(inp + row * NUM_FEATURES);
    float4 a = src[lane * 2];
    float4 b = src[lane * 2 + 1];
    float ss = a.x*a.x + a.y*a.y + a.z*a.z + a.w*a.w
             + b.x*b.x + b.y*b.y + b.z*b.z + b.w*b.w;
    ss = wave_sum32(ss);
    float sc = 1.0f / fmaxf(sqrtf(ss), EPSV);
    float4* dst = (float4*)(xn + row * NUM_FEATURES);
    a.x*=sc; a.y*=sc; a.z*=sc; a.w*=sc;
    b.x*=sc; b.y*=sc; b.z*=sc; b.w*=sc;
    dst[lane * 2]     = a;
    dst[lane * 2 + 1] = b;
    if (lane == 0) S[row] = 0.0f;
}

// ---------------------------------------------- 2. all-zero-row ("bad") mask
__global__ __launch_bounds__(256) void badmask_kernel(
    const float* __restrict__ lut, const float* __restrict__ cq,
    unsigned char* __restrict__ bad) {
    int row  = blockIdx.x * 8 + (threadIdx.x >> 5);
    int lane = threadIdx.x & 31;
    if (row >= TOTAL_COLS) return;
    const float* src = (row < NUM_PIDS) ? (lut + (size_t)row * NUM_FEATURES)
                                        : (cq + (size_t)(row - NUM_PIDS) * NUM_FEATURES);
    const float4* s4 = (const float4*)src;
    float4 a = s4[lane * 2];
    float4 b = s4[lane * 2 + 1];
    bool nz = (a.x != 0.f) | (a.y != 0.f) | (a.z != 0.f) | (a.w != 0.f)
            | (b.x != 0.f) | (b.y != 0.f) | (b.z != 0.f) | (b.w != 0.f);
    int any = __any(nz);
    if (lane == 0) bad[row] = any ? 0 : 1;
}

// -------------- 3. WMMA GEMM (bf16 -> f32) fused with masked exp-sum partial
__device__ __forceinline__ void epilogue_tile(
    v8f acc, int gcol, int mbase, const unsigned char* __restrict__ bad,
    const int* sLab, float* sRow, int half, int l16) {
    const bool valid = gcol < TOTAL_COLS;
    const bool isLut = gcol < NUM_PIDS;
    const unsigned char isBad = valid ? bad[gcol] : (unsigned char)1;
    #pragma unroll
    for (int e = 0; e < 8; ++e) {
        int m = mbase + half * 8 + e;          // C layout: lane N=l16, M=e+half*8
        float v = acc[e];
        if (isBad) v = (isLut && sLab[m] == gcol) ? 1.0f : -1.0f; // bad_pos override
        // logits bounded by +/-30: fixed shift by 30, no row-max pass needed
        float contrib = valid ? exp2f((OIM_SCALAR * v - OIM_SCALAR) * LOG2E) : 0.0f;
        contrib += __shfl_xor(contrib, 1, 32);
        contrib += __shfl_xor(contrib, 2, 32);
        contrib += __shfl_xor(contrib, 4, 32);
        contrib += __shfl_xor(contrib, 8, 32);
        if (l16 == 0) atomicAdd(&sRow[m], contrib);
    }
}

__global__ __launch_bounds__(256) void gemm_expsum_kernel(
    const float* __restrict__ xn, const float* __restrict__ lut,
    const float* __restrict__ cq, const unsigned char* __restrict__ bad,
    const int* __restrict__ label, float* __restrict__ S) {
    __shared__ __align__(16) __bf16 sA[TMB * LDSC];   // 17.0 KB
    __shared__ __align__(16) __bf16 sB[TNB * LDSC];   // 34.0 KB
    __shared__ float sRow[TMB];
    __shared__ int   sLab[TMB];

    const int tid     = threadIdx.x;
    const int rowbase = blockIdx.y * TMB;
    const int colbase = blockIdx.x * TNB;

    if (tid < TMB) { sRow[tid] = 0.0f; sLab[tid] = label[rowbase + tid]; }

    const int wave = tid >> 5;
    const int lane = tid & 31;
    const int wm   = wave >> 2;      // 0..1 : 32-row macro-tile
    const int wn   = wave & 3;       // 0..3 : 32-col macro-tile
    const int half = lane >> 4;
    const int l16  = lane & 15;

    v8f acc00 = {0.f,0.f,0.f,0.f,0.f,0.f,0.f,0.f};
    v8f acc01 = acc00, acc10 = acc00, acc11 = acc00;

    const int srow = tid >> 7;       // staging: 2 rows / iteration, 128 cols each
    const int scol = tid & 127;

    for (int chunk = 0; chunk < 2; ++chunk) {
        const int kbase = chunk * KC;

        // stage A chunk: 64 rows x 128 cols (f32 -> bf16), coalesced
        #pragma unroll 4
        for (int it = 0; it < TMB / 2; ++it) {
            int r = it * 2 + srow;
            sA[r * LDSC + scol] =
                (__bf16)xn[(size_t)(rowbase + r) * KDIM + kbase + scol];
        }
        // stage B chunk: 128 rows of concat(lut,cq); zero-fill OOB columns
        for (int it = 0; it < TNB / 2; ++it) {
            int r = it * 2 + srow;
            int col = colbase + r;
            float v = 0.0f;
            if (col < NUM_PIDS)        v = lut[(size_t)col * KDIM + kbase + scol];
            else if (col < TOTAL_COLS) v = cq[(size_t)(col - NUM_PIDS) * KDIM + kbase + scol];
            sB[r * LDSC + scol] = (__bf16)v;
        }
        __syncthreads();

        const __bf16* aR0 = &sA[(wm * 32 + l16) * LDSC];
        const __bf16* aR1 = aR0 + 16 * LDSC;
        const __bf16* bR0 = &sB[(wn * 32 + l16) * LDSC];
        const __bf16* bR1 = bR0 + 16 * LDSC;

        #pragma unroll
        for (int k0 = 0; k0 < KC; k0 += 32) {
            int cLo = k0 + half * 8;
            v16bf a0 = load_frag(aR0, cLo);
            v16bf a1 = load_frag(aR1, cLo);
            v16bf b0 = load_frag(bR0, cLo);
            v16bf b1 = load_frag(bR1, cLo);
            acc00 = __builtin_amdgcn_wmma_f32_16x16x32_bf16(false, a0, false, b0,
                        (short)0, acc00, false, false);
            acc01 = __builtin_amdgcn_wmma_f32_16x16x32_bf16(false, a0, false, b1,
                        (short)0, acc01, false, false);
            acc10 = __builtin_amdgcn_wmma_f32_16x16x32_bf16(false, a1, false, b0,
                        (short)0, acc10, false, false);
            acc11 = __builtin_amdgcn_wmma_f32_16x16x32_bf16(false, a1, false, b1,
                        (short)0, acc11, false, false);
        }
        __syncthreads();   // before restaging LDS
    }

    // epilogue: masked exp contributions, per-row partials into LDS
    const int gc0 = colbase + wn * 32 + l16;
    const int gc1 = gc0 + 16;
    const int mb0 = wm * 32;
    const int mb1 = mb0 + 16;
    epilogue_tile(acc00, gc0, mb0, bad, sLab, sRow, half, l16);
    epilogue_tile(acc01, gc1, mb0, bad, sLab, sRow, half, l16);
    epilogue_tile(acc10, gc0, mb1, bad, sLab, sRow, half, l16);
    epilogue_tile(acc11, gc1, mb1, bad, sLab, sRow, half, l16);
    __syncthreads();
    if (tid < TMB) atomicAdd(&S[rowbase + tid], sRow[tid]);
}

// ------------------------------------------- 4. logit at the label position
__global__ __launch_bounds__(256) void label_logit_kernel(
    const float* __restrict__ xn, const float* __restrict__ lut,
    const unsigned char* __restrict__ bad, const int* __restrict__ label,
    float* __restrict__ labL) {
    int row  = blockIdx.x * 8 + (threadIdx.x >> 5);
    int lane = threadIdx.x & 31;
    int y = label[row];
    float out = 0.0f;
    if (y < NUM_PIDS) {
        const float4* xr = (const float4*)(xn + (size_t)row * NUM_FEATURES);
        const float4* lr = (const float4*)(lut + (size_t)y * NUM_FEATURES);
        float4 xa = xr[lane*2], xb = xr[lane*2+1];
        float4 la = lr[lane*2], lb = lr[lane*2+1];
        float s = xa.x*la.x + xa.y*la.y + xa.z*la.z + xa.w*la.w
                + xb.x*lb.x + xb.y*lb.y + xb.z*lb.z + xb.w*lb.w;
        s = wave_sum32(s);
        float v = bad[y] ? 1.0f : s;
        out = OIM_SCALAR * v;
    }
    if (lane == 0) labL[row] = out;
}

// ------------------------------------------------------------- 5. final loss
__global__ __launch_bounds__(256) void loss_kernel(
    const float* __restrict__ S, const float* __restrict__ labL,
    const int* __restrict__ label, float* __restrict__ out) {
    __shared__ float red[256];
    float acc = 0.0f;
    for (int r = threadIdx.x; r < BATCH; r += 256)
        if (label[r] < NUM_PIDS)
            acc += OIM_SCALAR + logf(S[r]) - labL[r];   // logsumexp - label_logit
    red[threadIdx.x] = acc;
    __syncthreads();
    for (int s = 128; s > 0; s >>= 1) {
        if (threadIdx.x < s) red[threadIdx.x] += red[threadIdx.x + s];
        __syncthreads();
    }
    if (threadIdx.x == 0) out[0] = red[0] / (float)BATCH;
}

// --------------------- 6. momentum LUT update (sequential per pid, in order)
__global__ __launch_bounds__(256) void lut_update_kernel(
    const int* __restrict__ label, const float* __restrict__ lut,
    const float* __restrict__ xn, float* __restrict__ out_lut) {
    const int i = blockIdx.x;
    const int y = label[i];
    if (y >= NUM_PIDS) return;                 // uniform exit
    __shared__ int   sFound;
    __shared__ float sRed[8];
    const int t = threadIdx.x;
    if (t == 0) sFound = 0;
    __syncthreads();
    for (int j = t; j < i; j += 256)
        if (label[j] == y) sFound = 1;
    __syncthreads();
    if (sFound) return;                        // only first occurrence owns pid

    float v = lut[(size_t)y * NUM_FEATURES + t];
    for (int j = i; j < BATCH; ++j) {
        if (label[j] == y) {                   // uniform across block
            v = 0.5f * v + 0.5f * xn[(size_t)j * NUM_FEATURES + t];
            float ss = wave_sum32(v * v);
            if ((t & 31) == 0) sRed[t >> 5] = ss;
            __syncthreads();
            float tot = sRed[0]+sRed[1]+sRed[2]+sRed[3]+sRed[4]+sRed[5]+sRed[6]+sRed[7];
            v /= fmaxf(sqrtf(tot), EPSV);
            __syncthreads();                   // sRed reusable next match
        }
    }
    out_lut[(size_t)y * NUM_FEATURES + t] = v;
}

// ------------------------ 7. circular-queue update (rank = #unlabeled before)
__global__ __launch_bounds__(256) void cq_update_kernel(
    const int* __restrict__ label, const float* __restrict__ xn,
    float* __restrict__ out_cq) {
    const int i = blockIdx.x;
    if (label[i] < NUM_PIDS) return;           // labeled samples don't enqueue
    __shared__ int sCnt[8];
    const int t = threadIdx.x;
    int c = 0;
    for (int j = t; j < i; j += 256)
        if (label[j] >= NUM_PIDS) c++;
    for (int m = 1; m < 32; m <<= 1) c += __shfl_xor(c, m, 32);
    if ((t & 31) == 0) sCnt[t >> 5] = c;
    __syncthreads();
    int rank = sCnt[0]+sCnt[1]+sCnt[2]+sCnt[3]+sCnt[4]+sCnt[5]+sCnt[6]+sCnt[7];
    rank %= CQ_SIZE;                           // <=2048 unlabeled: never wraps
    out_cq[(size_t)rank * NUM_FEATURES + t] = xn[(size_t)i * NUM_FEATURES + t];
}

// ---------------------------------------------------------------- dispatcher
extern "C" void kernel_launch(void* const* d_in, const int* in_sizes, int n_in,
                              void* d_out, int out_size, void* d_ws, size_t ws_size,
                              hipStream_t stream) {
    const float* inputs = (const float*)d_in[0];
    const int*   label  = (const int*)d_in[1];
    const float* lut    = (const float*)d_in[2];
    const float* cq     = (const float*)d_in[3];

    float* out     = (float*)d_out;
    float* out_lut = out + 1;
    float* out_cq  = out_lut + (size_t)NUM_PIDS * NUM_FEATURES;

    // workspace layout (~2.2 MB)
    float* xn   = (float*)d_ws;                       // [2048 x 256]
    float* S    = xn + (size_t)BATCH * NUM_FEATURES;  // [2048]
    float* labL = S + BATCH;                          // [2048]
    unsigned char* bad = (unsigned char*)(labL + BATCH); // [55000]

    normalize_kernel<<<BATCH / 8, 256, 0, stream>>>(inputs, xn, S);
    badmask_kernel<<<(TOTAL_COLS + 7) / 8, 256, 0, stream>>>(lut, cq, bad);

    dim3 ggrid((TOTAL_COLS + TNB - 1) / TNB, BATCH / TMB);   // 430 x 32
    gemm_expsum_kernel<<<ggrid, 256, 0, stream>>>(xn, lut, cq, bad, label, S);

    label_logit_kernel<<<BATCH / 8, 256, 0, stream>>>(xn, lut, bad, label, labL);
    loss_kernel<<<1, 256, 0, stream>>>(S, labL, label, out);

    hipMemcpyAsync(out_lut, lut, (size_t)NUM_PIDS * NUM_FEATURES * sizeof(float),
                   hipMemcpyDeviceToDevice, stream);
    hipMemcpyAsync(out_cq, cq, (size_t)CQ_SIZE * NUM_FEATURES * sizeof(float),
                   hipMemcpyDeviceToDevice, stream);

    lut_update_kernel<<<BATCH, 256, 0, stream>>>(label, lut, xn, out_lut);
    cq_update_kernel<<<BATCH, 256, 0, stream>>>(label, xn, out_cq);
}
